// MultiHeadAttention_85779086836194
// MI455X (gfx1250) — matrix operand order repaired and checked by reading, not compiled, not run on hardware
//
#include <hip/hip_runtime.h>
#include <hip/hip_bf16.h>
#include <math.h>

#define Bsz 4
#define Ssz 1024
#define Dsz 1024
#define Hsz 16
#define DHsz 64

typedef _Float16 v16h __attribute__((ext_vector_type(16)));
typedef _Float16 v8h  __attribute__((ext_vector_type(8)));
typedef _Float16 v4h  __attribute__((ext_vector_type(4)));
typedef float    v8f  __attribute__((ext_vector_type(8)));

static __device__ __forceinline__ v16h cat16(v8h lo, v8h hi) {
  v16h r;
#pragma unroll
  for (int i = 0; i < 8; ++i) { r[i] = lo[i]; r[8 + i] = hi[i]; }
  return r;
}

static __device__ __forceinline__ v8f wmma_f16(v16h a, v16h b, v8f c) {
  return __builtin_amdgcn_wmma_f32_16x16x32_f16(false, a, false, b, (short)0, c,
                                                false, false);
}

// async copy of 16 bytes: global -> LDS, tracked by ASYNCcnt
static __device__ __forceinline__ void async_b128(unsigned lds_off,
                                                  const _Float16* g) {
  asm volatile("global_load_async_to_lds_b128 %0, %1, off"
               :: "v"(lds_off), "v"(g) : "memory");
}
static __device__ __forceinline__ void wait_async0() {
  asm volatile("s_wait_asynccnt 0x0" ::: "memory");
}

// ---------------------------------------------------------------------------
// Kernel 1: per-head QKV projection.
//   Q[bh][s][e] (f16 row-major), K[bh][t][e] (f16 row-major),
//   Vt[bh][e][t] (f16, V transposed).
// grid = (S/64, B*H), block = 128 (4 waves; each wave = one 16-row s tile)
// ---------------------------------------------------------------------------
__global__ void __launch_bounds__(128)
mha_qkv_kernel(const float* __restrict__ x,
               const float* __restrict__ Wq, const float* __restrict__ Wk,
               const float* __restrict__ Wv,
               const float* __restrict__ bq, const float* __restrict__ bk,
               const float* __restrict__ bv,
               _Float16* __restrict__ Qg, _Float16* __restrict__ Kg,
               _Float16* __restrict__ Vtg) {
  __shared__ _Float16 W_lds[3][64 * 64];
  __shared__ float    b_lds[3][64];
  __shared__ _Float16 x_lds[4][16 * 64];
  __shared__ _Float16 st_lds[4][16 * 16];

  const int tid  = threadIdx.x;
  const int wave = tid >> 5;
  const int lane = tid & 31;
  const int bh   = blockIdx.y;
  const int b    = bh >> 4;
  const int h    = bh & 15;
  const int s0   = blockIdx.x * 64 + wave * 16;

  {
    const float* wsrc[3] = {Wq + h * 4096, Wk + h * 4096, Wv + h * 4096};
#pragma unroll
    for (int m = 0; m < 3; ++m) {
      const float4* src = (const float4*)wsrc[m];
      for (int i = tid; i < 1024; i += 128) {
        float4 f = src[i];
        v4h hv = {(_Float16)f.x, (_Float16)f.y, (_Float16)f.z, (_Float16)f.w};
        *(v4h*)&W_lds[m][i * 4] = hv;
      }
    }
    if (tid < 64) {
      b_lds[0][tid] = bq[h * 64 + tid];
      b_lds[1][tid] = bk[h * 64 + tid];
      b_lds[2][tid] = bv[h * 64 + tid];
    }
  }

  {
    int row = lane >> 1;
    int cb  = (lane & 1) * 32;
    const float4* xg =
        (const float4*)(x + (size_t)(b * Ssz + s0 + row) * Dsz + h * 64 + cb);
#pragma unroll
    for (int i = 0; i < 8; ++i) {
      float4 f = xg[i];
      v4h hv = {(_Float16)f.x, (_Float16)f.y, (_Float16)f.z, (_Float16)f.w};
      *(v4h*)&x_lds[wave][row * 64 + cb + i * 4] = hv;
    }
  }
  __syncthreads();

  const int col   = lane & 15;
  const int mbase = (lane < 16) ? 0 : 8;

  auto loadA = [&](const _Float16* base, int stride, int kc) -> v16h {
    int m  = lane & 15;
    int ko = (lane < 16) ? 0 : 8;
    const _Float16* p = base + m * stride + kc + ko;
    return cat16(*(const v8h*)p, *(const v8h*)(p + 16));
  };
  auto loadB = [&](const _Float16* base, int stride, int n0, int kc) -> v16h {
    int n  = lane & 15;
    int kb = (lane < 16) ? 0 : 8;
    const _Float16* p = base + (n0 + n) * stride + kc + kb;
    return cat16(*(const v8h*)p, *(const v8h*)(p + 16));
  };

  v16h A0 = loadA(x_lds[wave], 64, 0);
  v16h A1 = loadA(x_lds[wave], 64, 32);

#pragma unroll
  for (int m = 0; m < 3; ++m) {
#pragma unroll
    for (int n0 = 0; n0 < 64; n0 += 16) {
      float bb = b_lds[m][n0 + col];
      v8f acc;
#pragma unroll
      for (int r = 0; r < 8; ++r) acc[r] = bb;
      acc = wmma_f16(A0, loadB(W_lds[m], 64, n0, 0), acc);
      acc = wmma_f16(A1, loadB(W_lds[m], 64, n0, 32), acc);

      if (m < 2) {
        _Float16* st = st_lds[wave];
#pragma unroll
        for (int r = 0; r < 8; ++r)
          st[(mbase + r) * 16 + col] = (_Float16)acc[r];
        asm volatile("s_wait_dscnt 0x0" ::: "memory");
        int rr = lane >> 1;
        int cc = (lane & 1) * 8;
        v8h vals = *(const v8h*)&st[rr * 16 + cc];
        _Float16* dst = (m == 0) ? Qg : Kg;
        *(v8h*)&dst[(size_t)(bh * Ssz + s0 + rr) * 64 + n0 + cc] = vals;
      } else {
        v8h pk;
#pragma unroll
        for (int r = 0; r < 8; ++r) pk[r] = (_Float16)acc[r];
        int e = n0 + col;
        *(v8h*)&Vtg[(size_t)(bh * 64 + e) * Ssz + s0 + mbase] = pk;
      }
    }
  }
}

// ---------------------------------------------------------------------------
// Kernel 2: flash attention per (b,h); 64-wide K/V tiles, async double buffer.
// grid = (S/128, B*H), block = 256 (8 waves; each wave = one 16-row q tile)
// ---------------------------------------------------------------------------
__global__ void __launch_bounds__(256)
mha_attn_kernel(const _Float16* __restrict__ Qg, const _Float16* __restrict__ Kg,
                const _Float16* __restrict__ Vtg, float* __restrict__ out) {
  __shared__ _Float16 Kt_lds[2][64 * 64];   // [buf][t][e]  8 KB each
  __shared__ _Float16 Vt_lds[2][64 * 64];   // [buf][e][t]  8 KB each
  __shared__ _Float16 P_lds[8][16 * 64];    // per-wave P staging, 16 KB

  const int tid  = threadIdx.x;
  const int wave = tid >> 5;
  const int lane = tid & 31;
  const int bh   = blockIdx.y;
  const int b    = bh >> 4;
  const int h    = bh & 15;
  const int s0   = blockIdx.x * 128 + wave * 16;

  const int col   = lane & 15;
  const int mbase = (lane < 16) ? 0 : 8;

  auto loadA = [&](const _Float16* base, int stride, int kc) -> v16h {
    int m  = lane & 15;
    int ko = (lane < 16) ? 0 : 8;
    const _Float16* p = base + m * stride + kc + ko;
    return cat16(*(const v8h*)p, *(const v8h*)(p + 16));
  };
  auto loadB = [&](const _Float16* base, int stride, int n0, int kc) -> v16h {
    int n  = lane & 15;
    int kb = (lane < 16) ? 0 : 8;
    const _Float16* p = base + (n0 + n) * stride + kc + kb;
    return cat16(*(const v8h*)p, *(const v8h*)(p + 16));
  };

  // per-thread async-copy of one 64x64 K tile + one 64x64 Vt tile
  const _Float16* kgbase = Kg + (size_t)bh * Ssz * 64;
  const _Float16* vgbase = Vtg + (size_t)bh * 64 * Ssz;
  const int ke = tid >> 2;            // row (t for K, e for V)
  const int kc4 = (tid & 3) * 16;     // 16-half chunk within the 64-half row
  auto prefetch_tile = [&](int t0, int buf) {
    unsigned kdst = (unsigned)(uintptr_t)&Kt_lds[buf][ke * 64 + kc4];
    const _Float16* ksrc = kgbase + (size_t)(t0 + ke) * 64 + kc4;
    async_b128(kdst, ksrc);
    async_b128(kdst + 16, ksrc + 8);
    unsigned vdst = (unsigned)(uintptr_t)&Vt_lds[buf][ke * 64 + kc4];
    const _Float16* vsrc = vgbase + (size_t)ke * Ssz + t0 + kc4;
    async_b128(vdst, vsrc);
    async_b128(vdst + 16, vsrc + 8);
  };

  // Q operands resident in VGPRs for the whole loop
  const _Float16* qb = Qg + (size_t)(bh * Ssz + s0) * 64;
  v16h QA0 = loadA(qb, 64, 0);
  v16h QA1 = loadA(qb, 64, 32);

  float mrow[8], lrow[8];
  v8f Oacc[4];
#pragma unroll
  for (int r = 0; r < 8; ++r) { mrow[r] = -INFINITY; lrow[r] = 0.0f; }
#pragma unroll
  for (int t = 0; t < 4; ++t)
#pragma unroll
    for (int r = 0; r < 8; ++r) Oacc[t][r] = 0.0f;

  prefetch_tile(0, 0);

  for (int j = 0; j < Ssz / 64; ++j) {
    const int buf = j & 1;
    wait_async0();       // this wave's async fills for buf have landed
    __syncthreads();     // ... and every other wave's too; prev compute done
    if (j + 1 < Ssz / 64) prefetch_tile((j + 1) * 64, buf ^ 1);

    const _Float16* Kt = Kt_lds[buf];
    const _Float16* Vt = Vt_lds[buf];

    // scores: 16 s-rows x 64 t-cols, f32 accum
    v8f Sacc[4];
#pragma unroll
    for (int q = 0; q < 4; ++q) {
#pragma unroll
      for (int r = 0; r < 8; ++r) Sacc[q][r] = 0.0f;
      Sacc[q] = wmma_f16(QA0, loadB(Kt, 64, q * 16, 0), Sacc[q]);
      Sacc[q] = wmma_f16(QA1, loadB(Kt, 64, q * 16, 32), Sacc[q]);
    }

    // online softmax over 64 columns at once
    _Float16* P = P_lds[wave];
#pragma unroll
    for (int r = 0; r < 8; ++r) {
      float sc[4];
#pragma unroll
      for (int q = 0; q < 4; ++q) sc[q] = Sacc[q][r] * 0.125f;  // 1/sqrt(64)
      float mx = fmaxf(fmaxf(sc[0], sc[1]), fmaxf(sc[2], sc[3]));
#pragma unroll
      for (int off = 1; off < 16; off <<= 1)
        mx = fmaxf(mx, __shfl_xor(mx, off, 32));
      float mnew = fmaxf(mrow[r], mx);
      float corr = __expf(mrow[r] - mnew);
      mrow[r] = mnew;
      float p0 = __expf(sc[0] - mnew);
      float p1 = __expf(sc[1] - mnew);
      float p2 = __expf(sc[2] - mnew);
      float p3 = __expf(sc[3] - mnew);
      float rs = (p0 + p1) + (p2 + p3);
#pragma unroll
      for (int off = 1; off < 16; off <<= 1) rs += __shfl_xor(rs, off, 32);
      lrow[r] = lrow[r] * corr + rs;
      Oacc[0][r] *= corr; Oacc[1][r] *= corr;
      Oacc[2][r] *= corr; Oacc[3][r] *= corr;
      P[(mbase + r) * 64 + col]      = (_Float16)p0;
      P[(mbase + r) * 64 + 16 + col] = (_Float16)p1;
      P[(mbase + r) * 64 + 32 + col] = (_Float16)p2;
      P[(mbase + r) * 64 + 48 + col] = (_Float16)p3;
    }
    asm volatile("s_wait_dscnt 0x0" ::: "memory");

    // O += P @ V
    v16h PA0 = loadA(P, 64, 0);
    v16h PA1 = loadA(P, 64, 32);
#pragma unroll
    for (int t = 0; t < 4; ++t) {
      Oacc[t] = wmma_f16(PA0, loadB(Vt, 64, t * 16, 0), Oacc[t]);
      Oacc[t] = wmma_f16(PA1, loadB(Vt, 64, t * 16, 32), Oacc[t]);
    }
  }

  // epilogue: out[b, s, h*64 + e] = O / l
#pragma unroll
  for (int t = 0; t < 4; ++t) {
    int e = t * 16 + col;
#pragma unroll
    for (int r = 0; r < 8; ++r) {
      float val = Oacc[t][r] / lrow[r];
      out[(size_t)(b * Ssz + s0 + mbase + r) * Dsz + h * 64 + e] = val;
    }
  }
}

// ---------------------------------------------------------------------------
extern "C" void kernel_launch(void* const* d_in, const int* in_sizes, int n_in,
                              void* d_out, int out_size, void* d_ws,
                              size_t ws_size, hipStream_t stream) {
  (void)in_sizes; (void)n_in; (void)out_size; (void)ws_size;
  const float* x  = (const float*)d_in[0];
  const float* Wq = (const float*)d_in[1];
  const float* Wk = (const float*)d_in[2];
  const float* Wv = (const float*)d_in[3];
  const float* bq = (const float*)d_in[4];
  const float* bk = (const float*)d_in[5];
  const float* bv = (const float*)d_in[6];
  float* out = (float*)d_out;

  const size_t elems = (size_t)Bsz * Hsz * Ssz * DHsz;  // 4M halfs = 8 MB each
  _Float16* Q  = (_Float16*)d_ws;
  _Float16* K  = Q + elems;
  _Float16* Vt = K + elems;

  mha_qkv_kernel<<<dim3(Ssz / 64, Bsz * Hsz), 128, 0, stream>>>(
      x, Wq, Wk, Wv, bq, bk, bv, Q, K, Vt);
  mha_attn_kernel<<<dim3(Ssz / 128, Bsz * Hsz), 256, 0, stream>>>(Q, K, Vt, out);
}
